// ReduceAFP_4234837754147
// MI455X (gfx1250) — matrix operand order repaired and checked by reading, not compiled, not run on hardware
//
#include <hip/hip_runtime.h>
#include <hip/hip_bf16.h>
#include <math.h>

typedef __attribute__((ext_vector_type(16))) _Float16 v16h;
typedef __attribute__((ext_vector_type(8)))  float    v8f;

#define DEV static __device__ __forceinline__

namespace {
constexpr int kNAtom = 200000, kEAtom = 400000;
constexpr int kNRg   = 40000,  kERg   = 80000;
constexpr int kNMol  = 8192;
constexpr int kC = 128, kInCh = 64, kEd = 16, kRgF = 18;
constexpr float kNeg = 0.01f;
constexpr float kNegInf = -3.0e38f;
}

DEV float act_apply(float v, int act) {
  if (act == 1) return v >= 0.f ? v : kNeg * v;   // leaky_relu
  if (act == 2) return v > 0.f ? v : 0.f;         // relu
  if (act == 3) return v > 0.f ? v : expm1f(v);   // elu
  return v;
}

DEV void atomic_max_f32(float* addr, float val) {
  unsigned int* u = (unsigned int*)addr;
  unsigned int old = __float_as_uint(*addr);
  while (__uint_as_float(old) < val) {
    unsigned int prev = atomicCAS(u, old, __float_as_uint(val));
    if (prev == old) break;
    old = prev;
  }
}

DEV v16h cvt16(float4 f0, float4 f1, float4 f2, float4 f3) {
  v16h r;
  r[0]=(_Float16)f0.x;  r[1]=(_Float16)f0.y;  r[2]=(_Float16)f0.z;  r[3]=(_Float16)f0.w;
  r[4]=(_Float16)f1.x;  r[5]=(_Float16)f1.y;  r[6]=(_Float16)f1.z;  r[7]=(_Float16)f1.w;
  r[8]=(_Float16)f2.x;  r[9]=(_Float16)f2.y;  r[10]=(_Float16)f2.z; r[11]=(_Float16)f2.w;
  r[12]=(_Float16)f3.x; r[13]=(_Float16)f3.y; r[14]=(_Float16)f3.z; r[15]=(_Float16)f3.w;
  return r;
}
// A fragment: elements 0..7 = p[0..7], elements 8..15 = p[16..23] (32B-aligned)
DEV v16h load_a16(const float* __restrict__ p) {
  return cvt16(*(const float4*)p, *(const float4*)(p + 4),
               *(const float4*)(p + 16), *(const float4*)(p + 20));
}
// B fragment: elements 0..15 = p[0..15] (16B-aligned)
DEV v16h load_b16(const float* __restrict__ p) {
  const float4* q = (const float4*)p;
  return cvt16(q[0], q[1], q[2], q[3]);
}
// B fragment, unaligned-safe scalar loads
DEV v16h load_b16_s(const float* __restrict__ p) {
  v16h r;
#pragma unroll
  for (int e = 0; e < 16; ++e) r[e] = (_Float16)p[e];
  return r;
}
DEV v8f do_wmma(v16h a, v16h b, v8f c) {
  return __builtin_amdgcn_wmma_f32_16x16x32_f16(false, a, false, b, (short)0, c,
                                                false, false);
}

// ---------------------------------------------------------------------------
// WMMA GEMM: Y[M,N] = act( concat(A1[gidx(m)], A2[m]) @ W[N,K].T + bias )
// Requirements (guaranteed by host): K1 % 32 == 0, N % 32 == 0.
// One wave computes a 16(M) x 32(N) tile: A fragment reused across 2 WMMAs.
// Row index clamped (not branched) -> no lane-divergent control flow; tail
// K-chunk (concat part) is a uniform loop with branchless clamped selects.
// ---------------------------------------------------------------------------
__global__ void __launch_bounds__(256)
afp_wmma_gemm(const float* __restrict__ A1, int K1,
              const float* __restrict__ A2, int K2,
              const int* __restrict__ gidx,
              const float* __restrict__ W,
              const float* __restrict__ bias,
              float* __restrict__ Y,
              int M, int N, int K, int act) {
  const int lane = threadIdx.x & 31;
  const int wave = threadIdx.x >> 5;
  const int m0 = (blockIdx.y * 8 + wave) * 16;
  const int n0 = blockIdx.x * 32;
  if (m0 >= M) return;

  const int l16 = lane & 15;
  const int hi  = lane >> 4;

  int arow = m0 + l16;
  if (arow >= M) arow = M - 1;                    // clamp, mask at store
  const size_t asrc = (size_t)(gidx ? gidx[arow] : arow);
  const float* __restrict__ ap = A1 + asrc * (size_t)K1;   // + k0 + hi*16

  const int bn = n0 + l16;                        // N%32==0 -> always valid
  const float* __restrict__ bp0 = W + (size_t)bn * (size_t)K;
  const float* __restrict__ bp1 = bp0 + (size_t)16 * (size_t)K;

  v8f acc0 = {}, acc1 = {};
  int k0 = 0;
  if ((K & 3) == 0) {                             // uniform: aligned-B variant
    for (; k0 < K1; k0 += 32) {
      const v16h a  = load_b16(ap + k0 + hi * 16);
      const v16h b0 = load_b16(bp0 + k0 + hi * 16);
      const v16h b1 = load_b16(bp1 + k0 + hi * 16);
      acc0 = do_wmma(a, b0, acc0);
      acc1 = do_wmma(a, b1, acc1);
    }
  } else {                                        // odd-K weights: scalar B
    for (; k0 < K1; k0 += 32) {
      const v16h a  = load_b16(ap + k0 + hi * 16);
      const v16h b0 = load_b16_s(bp0 + k0 + hi * 16);
      const v16h b1 = load_b16_s(bp1 + k0 + hi * 16);
      acc0 = do_wmma(a, b0, acc0);
      acc1 = do_wmma(a, b1, acc1);
    }
  }
  // ---- concat tail: k in [K1, K), A from A2[arow, k-K1]; branchless selects
  for (; k0 < K; k0 += 32) {
    v16h a, b0, b1;
#pragma unroll
    for (int e = 0; e < 16; ++e) {
      const int k  = k0 + hi * 16 + e;
      const int i2 = k - K1;                      // >= 0 here
      const int i2c = i2 < K2 ? i2 : (K2 - 1);
      const float va = A2[(size_t)arow * (size_t)K2 + i2c];
      a[e] = (_Float16)(i2 < K2 ? va : 0.f);
    }
#pragma unroll
    for (int e = 0; e < 16; ++e) {
      const int k  = k0 + hi * 16 + e;
      const int kc = k < K ? k : (K - 1);
      const float v0 = bp0[kc];
      const float v1 = bp1[kc];
      b0[e] = (_Float16)(k < K ? v0 : 0.f);
      b1[e] = (_Float16)(k < K ? v1 : 0.f);
    }
    acc0 = do_wmma(a, b0, acc0);
    acc1 = do_wmma(a, b1, acc1);
  }
  // ---- epilogue: D layout lane%16 = N col, M = r + (lane/16)*8; fused bias+act
  const float bia0 = bias ? bias[n0 + l16] : 0.f;
  const float bia1 = bias ? bias[n0 + 16 + l16] : 0.f;
#pragma unroll
  for (int r = 0; r < 8; ++r) {
    const int m = m0 + r + hi * 8;
    if (m < M) {
      float* yr = Y + (size_t)m * (size_t)N + n0 + l16;
      yr[0]  = act_apply(acc0[r] + bia0, act);
      yr[16] = act_apply(acc1[r] + bia1, act);
    }
  }
}

// y[row] = dot(X[row,:128], v) + (bias ? bias[0] : 0); one wave per row
__global__ void __launch_bounds__(256)
afp_rowdot(const float* __restrict__ X, const float* __restrict__ v,
           const float* __restrict__ bias, float* __restrict__ y, int M) {
  const int row = blockIdx.x * 8 + (threadIdx.x >> 5);
  const int lane = threadIdx.x & 31;
  if (row >= M) return;
  const float* xr = X + (size_t)row * kC;
  float s = xr[lane] * v[lane] + xr[lane + 32] * v[lane + 32] +
            xr[lane + 64] * v[lane + 64] + xr[lane + 96] * v[lane + 96];
#pragma unroll
  for (int off = 16; off > 0; off >>= 1) s += __shfl_xor(s, off, 32);
  if (lane == 0) y[row] = s + (bias ? bias[0] : 0.f);
}

// GATEConv logits: a = lrelu(dot(m_e[e], att_l) + xr[dst[e]]); seg-max
__global__ void __launch_bounds__(256)
afp_edge_att_gate(const float* __restrict__ m_e, const float* __restrict__ att_l,
                  const float* __restrict__ xr, const int* __restrict__ dst,
                  float* __restrict__ a_raw, float* __restrict__ smax, int E) {
  const int e = blockIdx.x * 8 + (threadIdx.x >> 5);
  const int lane = threadIdx.x & 31;
  if (e >= E) return;
  const float* mr = m_e + (size_t)e * kC;
  float s = mr[lane] * att_l[lane] + mr[lane + 32] * att_l[lane + 32] +
            mr[lane + 64] * att_l[lane + 64] + mr[lane + 96] * att_l[lane + 96];
#pragma unroll
  for (int off = 16; off > 0; off >>= 1) s += __shfl_xor(s, off, 32);
  if (lane == 0) {
    const int d = dst[e];
    float a = s + xr[d];
    a = a >= 0.f ? a : kNeg * a;
    a_raw[e] = a;
    atomic_max_f32(smax + d, a);
  }
}

// GATConv logits: a = lrelu(es[src] + ed[dst]); seg-max  (null idx = identity)
__global__ void afp_edge_att_gat(const float* __restrict__ es, const float* __restrict__ ed,
                                 const int* __restrict__ src, const int* __restrict__ dst,
                                 float* __restrict__ a_raw, float* __restrict__ smax, int E) {
  const int e = blockIdx.x * blockDim.x + threadIdx.x;
  if (e >= E) return;
  const int s = src ? src[e] : e;
  const int d = dst ? dst[e] : e;
  float a = es[s] + ed[d];
  a = a >= 0.f ? a : kNeg * a;
  a_raw[e] = a;
  atomic_max_f32(smax + d, a);
}

// e_val = exp(a - segmax[dst]); seg-sum
__global__ void afp_edge_exp(const float* __restrict__ a_raw, const float* __restrict__ smax,
                             const int* __restrict__ dst, float* __restrict__ eval_,
                             float* __restrict__ ssum, int E) {
  const int e = blockIdx.x * blockDim.x + threadIdx.x;
  if (e >= E) return;
  const int d = dst ? dst[e] : e;
  const float v = expf(a_raw[e] - smax[d]);
  eval_[e] = v;
  atomicAdd(ssum + d, v);
}

// out[dst[e], :] += V[vidx(e), :] * (eval[e]/ssum[dst[e]] or 1); 2 edges / block
__global__ void __launch_bounds__(256)
afp_scatter_add(const float* __restrict__ V, const int* __restrict__ vidx,
                const int* __restrict__ dst, const float* __restrict__ eval_,
                const float* __restrict__ ssum, float* __restrict__ out, int E) {
  const int c = threadIdx.x & 127;
  const long long e = blockIdx.x * 2LL + (threadIdx.x >> 7);
  if (e >= E) return;
  const long long row = vidx ? (long long)vidx[e] : e;
  const long long d   = dst  ? (long long)dst[e]  : e;
  const float w = eval_ ? eval_[e] / ssum[d] : 1.f;
  atomicAdd(out + d * kC + c, V[row * kC + c] * w);
}

__global__ void afp_bias_act(float* __restrict__ X, const float* __restrict__ bias,
                             long long M, int act) {
  const long long total = M * kC;
  const long long i = blockIdx.x * (long long)blockDim.x + threadIdx.x;
  if (i >= total) return;
  const int c = (int)(i & (kC - 1));
  float v = X[i] + (bias ? bias[c] : 0.f);
  X[i] = act_apply(v, act);
}

// h <- relu(GRU(gi, gh, h)), in place over the state buffer
__global__ void afp_gru_combine(const float* __restrict__ gi, const float* __restrict__ gh,
                                float* __restrict__ h, long long M) {
  const long long total = M * kC;
  const long long i = blockIdx.x * (long long)blockDim.x + threadIdx.x;
  if (i >= total) return;
  const long long row = i >> 7;
  const int c = (int)(i & (kC - 1));
  const float* gir = gi + row * 3 * kC;
  const float* ghr = gh + row * 3 * kC;
  const float r = 1.f / (1.f + expf(-(gir[c] + ghr[c])));
  const float z = 1.f / (1.f + expf(-(gir[kC + c] + ghr[kC + c])));
  const float n = tanhf(gir[2 * kC + c] + r * ghr[2 * kC + c]);
  const float v = (1.f - z) * n + z * h[i];
  h[i] = v > 0.f ? v : 0.f;
}

__global__ void afp_fill(float* __restrict__ p, float v, long long n) {
  const long long i = blockIdx.x * (long long)blockDim.x + threadIdx.x;
  if (i < n) p[i] = v;
}

// ---------------------------------------------------------------------------
// Host-side helpers
// ---------------------------------------------------------------------------
static inline unsigned nblk(long long n, long long per) {
  long long b = (n + per - 1) / per;
  return (unsigned)(b < 1 ? 1 : b);
}

static void h_gemm(hipStream_t s, const float* A1, int K1, const float* A2, int K2,
                   const int* gidx, const float* W, const float* bias, float* Y,
                   int M, int N, int K, int act) {
  dim3 grid(N / 32, nblk(M, 128));
  afp_wmma_gemm<<<grid, 256, 0, s>>>(A1, K1, A2, K2, gidx, W, bias, Y, M, N, K, act);
}
static void h_rowdot(hipStream_t s, const float* X, const float* v, const float* bias,
                     float* y, int M) {
  afp_rowdot<<<nblk(M, 8), 256, 0, s>>>(X, v, bias, y, M);
}
static void h_fill(hipStream_t s, float* p, float v, long long n) {
  afp_fill<<<nblk(n, 256), 256, 0, s>>>(p, v, n);
}
static void h_scatter(hipStream_t s, const float* V, const int* vidx, const int* dst,
                      const float* eval_, const float* ssum, float* out, int E) {
  afp_scatter_add<<<nblk(E, 2), 256, 0, s>>>(V, vidx, dst, eval_, ssum, out, E);
}
static void h_bias_act(hipStream_t s, float* X, const float* bias, long long M, int act) {
  afp_bias_act<<<nblk(M * kC, 256), 256, 0, s>>>(X, bias, M, act);
}

// GATEConv: out = elu(scatter(softmax_dst(lrelu(m@att_l + xr[dst])) * (m@W2.T)) + b)
static void run_gate(hipStream_t s, const float* xn, int Nn, const float* eat, int K2,
                     const int* src, const int* dst, int E,
                     const float* W1, const float* W2, const float* att_l,
                     const float* att_r, const float* b,
                     float* m_e, float* y2, float* xr, float* a_raw, float* eval_,
                     float* smax, float* ssum, float* out) {
  h_gemm(s, xn, kC, eat, K2, src, W1, nullptr, m_e, E, kC, kC + K2, /*lrelu*/1);
  h_rowdot(s, xn, att_r, nullptr, xr, Nn);
  h_fill(s, smax, kNegInf, Nn);
  afp_edge_att_gate<<<nblk(E, 8), 256, 0, s>>>(m_e, att_l, xr, dst, a_raw, smax, E);
  h_fill(s, ssum, 0.f, Nn);
  afp_edge_exp<<<nblk(E, 256), 256, 0, s>>>(a_raw, smax, dst, eval_, ssum, E);
  h_gemm(s, m_e, kC, nullptr, 0, nullptr, W2, nullptr, y2, E, kC, kC, 0);
  h_fill(s, out, 0.f, (long long)Nn * kC);
  h_scatter(s, y2, nullptr, dst, eval_, ssum, out, E);
  h_bias_act(s, out, b, Nn, /*elu*/3);
}

// GATConv (heads=1): out = elu(scatter(softmax_dst(lrelu(es[src]+ed[dst])) * hs[src]) + b)
static void run_gat(hipStream_t s, const float* xsrc, const float* xdst, int Ns, int Nd,
                    const int* src, const int* dst, int E,
                    const float* W, const float* a_src, const float* a_dst, const float* b,
                    float* hs, float* hd_buf, float* es, float* ed,
                    float* a_raw, float* eval_, float* smax, float* ssum, float* out) {
  h_gemm(s, xsrc, kC, nullptr, 0, nullptr, W, nullptr, hs, Ns, kC, kC, 0);
  const float* hd = hs;
  if (xdst != xsrc) {
    h_gemm(s, xdst, kC, nullptr, 0, nullptr, W, nullptr, hd_buf, Nd, kC, kC, 0);
    hd = hd_buf;
  }
  h_rowdot(s, hs, a_src, nullptr, es, Ns);
  h_rowdot(s, hd, a_dst, nullptr, ed, Nd);
  h_fill(s, smax, kNegInf, Nd);
  afp_edge_att_gat<<<nblk(E, 256), 256, 0, s>>>(es, ed, src, dst, a_raw, smax, E);
  h_fill(s, ssum, 0.f, Nd);
  afp_edge_exp<<<nblk(E, 256), 256, 0, s>>>(a_raw, smax, dst, eval_, ssum, E);
  h_fill(s, out, 0.f, (long long)Nd * kC);
  h_scatter(s, hs, src, dst, eval_, ssum, out, E);
  h_bias_act(s, out, b, Nd, /*elu*/3);
}

// hstate <- relu(GRU(hin, hstate))
static void run_gru(hipStream_t s, const float* hin, float* hstate,
                    const float* wih, const float* whh, const float* bih, const float* bhh,
                    float* gi, float* gh, int M) {
  h_gemm(s, hin,    kC, nullptr, 0, nullptr, wih, bih, gi, M, 3 * kC, kC, 0);
  h_gemm(s, hstate, kC, nullptr, 0, nullptr, whh, bhh, gh, M, 3 * kC, kC, 0);
  afp_gru_combine<<<nblk((long long)M * kC, 256), 256, 0, s>>>(gi, gh, hstate, M);
}

// ---------------------------------------------------------------------------
extern "C" void kernel_launch(void* const* d_in, const int* in_sizes, int n_in,
                              void* d_out, int out_size, void* d_ws, size_t ws_size,
                              hipStream_t stream) {
  (void)in_sizes; (void)n_in; (void)out_size; (void)ws_size;
  auto F = [&](int i) { return (const float*)d_in[i]; };

  // ---- inputs (setup_inputs insertion order, params flattened in insertion order)
  const float* x        = F(0);   // [N_ATOM, 64]
  const float* eattr    = F(1);   // [E_ATOM, 16]
  const float* rg_x     = F(2);   // [N_RG, 18]
  const float* rg_eattr = F(3);   // [E_RG, 1]
  const float *lin1W = F(4), *lin1b = F(5);
  const float *gaL = F(6), *gaR = F(7), *gaW1 = F(8), *gaW2 = F(9), *gaB = F(10);
  const float *g0wih = F(11), *g0whh = F(12), *g0bih = F(13), *g0bhh = F(14);
  const float *gtW = F(15), *gtAs = F(16), *gtAd = F(17), *gtB = F(18);
  const float *g1wih = F(19), *g1whh = F(20), *g1bih = F(21), *g1bhh = F(22);
  const float *pcW = F(23), *pcAs = F(24), *pcAd = F(25), *pcB = F(26);
  const float *pgwih = F(27), *pgwhh = F(28), *pgbih = F(29), *pgbhh = F(30);
  const float *lrgW = F(31), *lrgb = F(32);
  const float *grL = F(33), *grR = F(34), *grW1 = F(35), *grW2 = F(36), *grB = F(37);
  const float *r0wih = F(38), *r0whh = F(39), *r0bih = F(40), *r0bhh = F(41);
  const float *rgtW = F(42), *rgtAs = F(43), *rgtAd = F(44), *rgtB = F(45);
  const float *r1wih = F(46), *r1whh = F(47), *r1bih = F(48), *r1bhh = F(49);
  const float *mcW = F(50), *mcAs = F(51), *mcAd = F(52), *mcB = F(53);
  const float *mgwih = F(54), *mgwhh = F(55), *mgbih = F(56), *mgbhh = F(57);
  const float *lin2W = F(58), *lin2b = F(59);
  const int* ei    = (const int*)d_in[60];
  const int* ei_src = ei;            const int* ei_dst = ei + kEAtom;
  const int* pool  = (const int*)d_in[61];
  const int* tid   = pool;           const int* sid    = pool + kNAtom;
  const int* rgei  = (const int*)d_in[62];
  const int* rgs   = rgei;           const int* rgd    = rgei + kERg;
  const int* rg_batch = (const int*)d_in[63];

  // ---- workspace layout (floats)
  float* ws = (float*)d_ws;
  size_t off = 0;
  auto alloc = [&](size_t n) { float* p = ws + off; off += n; return p; };
  float* XA    = alloc((size_t)kNAtom * kC);          // atom state
  float* Hb    = alloc((size_t)kNAtom * kC);          // conv output h
  float* ARENA = alloc((size_t)2 * kNAtom * 3 * kC);  // gi/gh or m_e/y2
  float* A_RAW = alloc(kEAtom);
  float* EVAL  = alloc(kEAtom);
  float* SMAX  = alloc(kNAtom);
  float* SSUM  = alloc(kNAtom);
  float* ES    = alloc(kNAtom);                       // also xr
  float* EDv   = alloc(kNAtom);
  float* RG_SUM = alloc((size_t)kNRg * kC);           // rg_out_ (fixed pooled sum)
  float* RG_OUT = alloc((size_t)kNRg * kC);           // rg GRU state
  float* RG2    = alloc((size_t)kNRg * kC);           // rg features after lin_rg
  float* HRG    = alloc((size_t)kNRg * kC);
  float* HMOL   = alloc((size_t)kNMol * kC);
  float* GI = ARENA;
  float* GH = ARENA + (size_t)kNAtom * 3 * kC;
  float* ME = ARENA;                                  // [E_ATOM, C]
  float* Y2 = ARENA + (size_t)kEAtom * kC;            // [E_ATOM, C]

  float* PRED = (float*)d_out;                        // [8192]
  float* MOL  = (float*)d_out + kNMol;                // [8192, 128] final 'out'

  // ================= atom passing =================
  h_gemm(stream, x, kInCh, nullptr, 0, nullptr, lin1W, lin1b, XA, kNAtom, kC, kInCh, 1);
  run_gate(stream, XA, kNAtom, eattr, kEd, ei_src, ei_dst, kEAtom,
           gaW1, gaW2, gaL, gaR, gaB, ME, Y2, ES, A_RAW, EVAL, SMAX, SSUM, Hb);
  run_gru(stream, Hb, XA, g0wih, g0whh, g0bih, g0bhh, GI, GH, kNAtom);
  run_gat(stream, XA, XA, kNAtom, kNAtom, ei_src, ei_dst, kEAtom,
          gtW, gtAs, gtAd, gtB, ME, Y2, ES, EDv, A_RAW, EVAL, SMAX, SSUM, Hb);
  run_gru(stream, Hb, XA, g1wih, g1whh, g1bih, g1bhh, GI, GH, kNAtom);
  // XA now == atom 'out'

  // ================= atom -> reduced-graph pooling =================
  h_fill(stream, RG_SUM, 0.f, (long long)kNRg * kC);
  h_scatter(stream, XA, sid, tid, nullptr, nullptr, RG_SUM, kNAtom);      // rg_out_
  // h = elu(GATConv(out, rg_out_, pool_ei)) is loop-invariant: compute once
  run_gat(stream, XA, RG_SUM, kNAtom, kNRg, sid, tid, kNAtom,
          pcW, pcAs, pcAd, pcB, ME, Y2, ES, EDv, A_RAW, EVAL, SMAX, SSUM, HRG);
  hipMemcpyAsync(RG_OUT, RG_SUM, (size_t)kNRg * kC * sizeof(float),
                 hipMemcpyDeviceToDevice, stream);
  run_gru(stream, HRG, RG_OUT, pgwih, pgwhh, pgbih, pgbhh, GI, GH, kNRg);
  run_gru(stream, HRG, RG_OUT, pgwih, pgwhh, pgbih, pgbhh, GI, GH, kNRg);

  // ================= reduced-graph passing =================
  h_gemm(stream, RG_OUT, kC, rg_x, kRgF, nullptr, lrgW, lrgb, RG2, kNRg, kC, kC + kRgF, 0);
  run_gate(stream, RG2, kNRg, rg_eattr, 1, rgs, rgd, kERg,
           grW1, grW2, grL, grR, grB, ME, Y2, ES, A_RAW, EVAL, SMAX, SSUM, HRG);
  run_gru(stream, HRG, RG2, r0wih, r0whh, r0bih, r0bhh, GI, GH, kNRg);
  run_gat(stream, RG2, RG2, kNRg, kNRg, rgs, rgd, kERg,
          rgtW, rgtAs, rgtAd, rgtB, ME, Y2, ES, EDv, A_RAW, EVAL, SMAX, SSUM, HRG);
  run_gru(stream, HRG, RG2, r1wih, r1whh, r1bih, r1bhh, GI, GH, kNRg);
  // RG2 now == final rg_out

  // ================= molecule passing =================
  h_fill(stream, MOL, 0.f, (long long)kNMol * kC);
  h_scatter(stream, RG2, nullptr, rg_batch, nullptr, nullptr, MOL, kNRg);
  h_bias_act(stream, MOL, nullptr, kNMol, /*relu*/2);
  for (int it = 0; it < 2; ++it) {
    // mol_ei: src = arange(N_RG) (identity), dst = rg_batch
    run_gat(stream, RG2, MOL, kNRg, kNMol, nullptr, rg_batch, kNRg,
            mcW, mcAs, mcAd, mcB, ME, Y2, ES, EDv, A_RAW, EVAL, SMAX, SSUM, HMOL);
    run_gru(stream, HMOL, MOL, mgwih, mgwhh, mgbih, mgbhh, GI, GH, kNMol);
  }
  // pred = out @ lin2.W.T + b
  h_rowdot(stream, MOL, lin2W, lin2b, PRED, kNMol);
}
